// transformer_block_5488968204586
// MI455X (gfx1250) — matrix operand order repaired and checked by reading, not compiled, not run on hardware
//
#include <hip/hip_runtime.h>
#include <cstddef>
#include <cstdint>

// ---------------------------------------------------------------------------
// Problem constants (reference: B=2, S=2048, D=1024, H=16, hd=64)
// ---------------------------------------------------------------------------
#define DMODEL 1024
#define NHEAD  16
#define SEQ    2048
#define HD     64
#define BATCH  2
#define ROWS   (BATCH * SEQ)          // 4096 tokens
#define BH     (BATCH * NHEAD)        // 32 batched heads
#define EPS    1e-5f

// GEMM tiling: block tile 128x64, K step 32 (one bf16 WMMA K), 256 thr = 8 waves
#define BM 128
#define BN 64
#define BK 32

typedef __attribute__((ext_vector_type(16))) __bf16 v16bf;
typedef __attribute__((ext_vector_type(8)))  __bf16 v8bf;
typedef __attribute__((ext_vector_type(8)))  float  v8f;

// ---------------------------------------------------------------------------
// CDNA5 async global->LDS copy (ASYNCcnt path, bypasses VGPRs).
// Each lane copies 16 bytes. LDS operand is the 32-bit LDS offset (low bits of
// the generic pointer), global operand is the 64-bit address.
// ---------------------------------------------------------------------------
__device__ __forceinline__ void async_b128_to_lds(const __bf16* g, __bf16* l) {
  uint32_t lds = (uint32_t)(size_t)l;
  uint64_t ga  = (uint64_t)(size_t)g;
  asm volatile("global_load_async_to_lds_b128 %0, %1, off"
               :: "v"(lds), "v"(ga) : "memory");
}
__device__ __forceinline__ void wait_asynccnt0() {
  asm volatile("s_wait_asynccnt 0" ::: "memory");
}

// ---------------------------------------------------------------------------
// fp32 -> bf16 cast (one-time weight conversion + misc)
// ---------------------------------------------------------------------------
__global__ __launch_bounds__(256) void cast_f32_to_bf16(const float* __restrict__ in,
                                                        __bf16* __restrict__ out, int n) {
  int i = (blockIdx.x * 256 + threadIdx.x) * 4;
  if (i < n) {
    float4 f = *reinterpret_cast<const float4*>(in + i);
    out[i + 0] = (__bf16)f.x;
    out[i + 1] = (__bf16)f.y;
    out[i + 2] = (__bf16)f.z;
    out[i + 3] = (__bf16)f.w;
  }
}

// ---------------------------------------------------------------------------
// LayerNorm (one 256-thread block per row of D=1024) -> bf16
// ---------------------------------------------------------------------------
__global__ __launch_bounds__(256) void layernorm_to_bf16(const float* __restrict__ x,
                                                         const float* __restrict__ gam,
                                                         const float* __restrict__ bet,
                                                         __bf16* __restrict__ out) {
  __shared__ float red[256];
  const int row = blockIdx.x;
  const int t = threadIdx.x;
  const float* xr = x + (size_t)row * DMODEL;
  float4 v = reinterpret_cast<const float4*>(xr)[t];

  red[t] = v.x + v.y + v.z + v.w;
  __syncthreads();
  for (int off = 128; off > 0; off >>= 1) {
    if (t < off) red[t] += red[t + off];
    __syncthreads();
  }
  const float mean = red[0] * (1.0f / DMODEL);
  __syncthreads();

  float dx = v.x - mean, dy = v.y - mean, dz = v.z - mean, dw = v.w - mean;
  red[t] = dx * dx + dy * dy + dz * dz + dw * dw;
  __syncthreads();
  for (int off = 128; off > 0; off >>= 1) {
    if (t < off) red[t] += red[t + off];
    __syncthreads();
  }
  const float rstd = rsqrtf(red[0] * (1.0f / DMODEL) + EPS);

  float4 g = reinterpret_cast<const float4*>(gam)[t];
  float4 b = reinterpret_cast<const float4*>(bet)[t];
  __bf16* o = out + (size_t)row * DMODEL + t * 4;
  o[0] = (__bf16)(dx * rstd * g.x + b.x);
  o[1] = (__bf16)(dy * rstd * g.y + b.y);
  o[2] = (__bf16)(dz * rstd * g.z + b.z);
  o[3] = (__bf16)(dw * rstd * g.w + b.w);
}

// ---------------------------------------------------------------------------
// Row softmax over length-2048 rows (in place, fp32). Masked entries (-1e10)
// underflow to exactly 0 after exp, matching the reference.
// ---------------------------------------------------------------------------
__global__ __launch_bounds__(256) void softmax_rows(float* __restrict__ w) {
  __shared__ float red[256];
  const int t = threadIdx.x;
  float* p = w + (size_t)blockIdx.x * SEQ;
  float4 a = reinterpret_cast<float4*>(p)[t * 2];
  float4 b = reinterpret_cast<float4*>(p)[t * 2 + 1];

  float mx = fmaxf(fmaxf(fmaxf(a.x, a.y), fmaxf(a.z, a.w)),
                   fmaxf(fmaxf(b.x, b.y), fmaxf(b.z, b.w)));
  red[t] = mx;
  __syncthreads();
  for (int off = 128; off > 0; off >>= 1) {
    if (t < off) red[t] = fmaxf(red[t], red[t + off]);
    __syncthreads();
  }
  const float rowmax = red[0];
  __syncthreads();

  a.x = __expf(a.x - rowmax); a.y = __expf(a.y - rowmax);
  a.z = __expf(a.z - rowmax); a.w = __expf(a.w - rowmax);
  b.x = __expf(b.x - rowmax); b.y = __expf(b.y - rowmax);
  b.z = __expf(b.z - rowmax); b.w = __expf(b.w - rowmax);

  red[t] = a.x + a.y + a.z + a.w + b.x + b.y + b.z + b.w;
  __syncthreads();
  for (int off = 128; off > 0; off >>= 1) {
    if (t < off) red[t] += red[t + off];
    __syncthreads();
  }
  const float inv = 1.0f / red[0];
  a.x *= inv; a.y *= inv; a.z *= inv; a.w *= inv;
  b.x *= inv; b.y *= inv; b.z *= inv; b.w *= inv;
  reinterpret_cast<float4*>(p)[t * 2] = a;
  reinterpret_cast<float4*>(p)[t * 2 + 1] = b;
}

// ---------------------------------------------------------------------------
// A-tile staging: bf16 source -> async DMA straight into LDS (no VGPRs);
// fp32 source (the softmaxed attention matrix) -> convert in VGPRs.
// ---------------------------------------------------------------------------
__device__ __forceinline__ void stage_a_row(const __bf16* __restrict__ src, __bf16* dst) {
  async_b128_to_lds(src, dst);           // 8 halfs
  async_b128_to_lds(src + 8, dst + 8);   // 8 halfs
}
__device__ __forceinline__ void stage_a_row(const float* __restrict__ src, __bf16* dst) {
#pragma unroll
  for (int j = 0; j < 16; j += 4) {
    float4 f = *reinterpret_cast<const float4*>(src + j);
    dst[j + 0] = (__bf16)f.x;
    dst[j + 1] = (__bf16)f.y;
    dst[j + 2] = (__bf16)f.z;
    dst[j + 3] = (__bf16)f.w;
  }
}

// ---------------------------------------------------------------------------
// Tiled bf16 WMMA GEMM: C = A @ B, epilogue functor handles bias/mask/residual
// and the final store. grid = (N/BN, M/BM, Z). 256 threads = 8 waves in a
// 4x2 grid; each wave owns a 32x32 tile = 2x2 v_wmma_f32_16x16x32_bf16 frags.
//
// Pipeline per K-step s (double-buffered LDS, buffer p = s&1):
//   - issue async-A DMA for step s+1 into buffer p^1 (fire-and-forget)
//   - fragment ds_load_b128 from buffer p, 4x WMMA
//   - ds_store the B row already held in VGPRs (loaded during step s-1)
//     into buffer p^1  -> global-load latency hidden behind the WMMAs
//   - issue the global_load of B for step s+2 into VGPRs
//   - s_wait_asynccnt 0 ; barrier
//
// Fragment layouts per CDNA5 ISA 7.12.2:
//   A 16x32 bf16: lane<16 holds row M=lane, K {0..7,16..23}; lane>=16 same row,
//                 K {8..15,24..31}  -> two 16B LDS loads + shuffle.
//   B 32x16 bf16: lane<16 holds col N=lane, K {0..7,16..23}; lane>=16 K {8..15,24..31}.
//                 B staged N-major in LDS so each frag is two 16B loads.
//   C/D 16x16 f32: VGPR i -> (M = (lane>>4)*8 + i, N = lane&15).
// ---------------------------------------------------------------------------
template <typename AT, typename Epi>
__global__ __launch_bounds__(256)
void gemm_bf16_wmma(const AT* __restrict__ Ag, int lda, long long strideA,
                    const __bf16* __restrict__ Bg, int ldb, long long strideB,
                    int K, Epi epi) {
  __shared__ __bf16 As[2][BM][BK];   // row-major M x K tiles (double buffered)
  __shared__ __bf16 Bst[2][BN][BK];  // N-major (transposed) K x N tiles

  const int t = threadIdx.x;
  const int lane = t & 31;
  const int wave = t >> 5;
  const int wm = (wave >> 1) * 32;  // wave row offset: 0,32,64,96
  const int wn = (wave & 1) * 32;   // wave col offset: 0,32
  const int m0 = blockIdx.y * BM;
  const int n0 = blockIdx.x * BN;
  const int z = blockIdx.z;
  Ag += (size_t)z * strideA;
  Bg += (size_t)z * strideB;

  v8f acc[2][2];
#pragma unroll
  for (int i = 0; i < 2; ++i)
#pragma unroll
    for (int j = 0; j < 2; ++j)
#pragma unroll
      for (int e = 0; e < 8; ++e) acc[i][j][e] = 0.0f;

  const int ar = t >> 1;             // A stage: 2 threads/row, 16 elems each
  const int ac = (t & 1) * 16;
  const int br = t >> 3;             // B stage: 8 threads/row, 8 elems each
  const int bc = (t & 7) * 8;

  const AT* aptr = Ag + (size_t)(m0 + ar) * lda + ac;       // + k0
  const __bf16* bptr = Bg + (size_t)br * ldb + n0 + bc;     // + k0*ldb

  auto load_b = [&](int k0) -> v8bf {
    return *reinterpret_cast<const v8bf*>(bptr + (size_t)k0 * ldb);
  };
  auto store_b = [&](int buf, v8bf rowv) {
#pragma unroll
    for (int j = 0; j < 8; ++j) Bst[buf][bc + j][br] = rowv[j];
  };

  const int nsteps = K / BK;

  // Prologue: tile 0 fully staged; B for tile 1 pre-loaded into VGPRs.
  v8bf bnext = load_b(0);
  stage_a_row(aptr, &As[0][ar][ac]);
  store_b(0, bnext);
  if (nsteps > 1) bnext = load_b(BK);
  wait_asynccnt0();
  __syncthreads();

  for (int s = 0; s < nsteps; ++s) {
    const int p = s & 1;
    if (s + 1 < nsteps) stage_a_row(aptr + (s + 1) * BK, &As[p ^ 1][ar][ac]);
    if (s + 2 < nsteps) {  // warm L2 two tiles ahead (global_prefetch_b8)
      __builtin_prefetch(aptr + (s + 2) * BK, 0, 1);
      __builtin_prefetch(bptr + (size_t)(s + 2) * BK * ldb, 0, 1);
    }

    const int arow = wm + (lane & 15);
    const int akb = (lane >> 4) * 8;
    v16bf afrag[2];
#pragma unroll
    for (int i = 0; i < 2; ++i) {
      const __bf16* pp = &As[p][arow + i * 16][0];
      v8bf lo = *reinterpret_cast<const v8bf*>(pp + akb);
      v8bf hi = *reinterpret_cast<const v8bf*>(pp + 16 + akb);
      afrag[i] = __builtin_shufflevector(lo, hi, 0, 1, 2, 3, 4, 5, 6, 7,
                                         8, 9, 10, 11, 12, 13, 14, 15);
    }
    const int bcol = wn + (lane & 15);
    const int bkb = (lane >> 4) * 8;
    v16bf bfrag[2];
#pragma unroll
    for (int j = 0; j < 2; ++j) {
      const __bf16* pp = &Bst[p][bcol + j * 16][0];
      v8bf lo = *reinterpret_cast<const v8bf*>(pp + bkb);
      v8bf hi = *reinterpret_cast<const v8bf*>(pp + 16 + bkb);
      bfrag[j] = __builtin_shufflevector(lo, hi, 0, 1, 2, 3, 4, 5, 6, 7,
                                         8, 9, 10, 11, 12, 13, 14, 15);
    }
#pragma unroll
    for (int i = 0; i < 2; ++i)
#pragma unroll
      for (int j = 0; j < 2; ++j)
        acc[i][j] = __builtin_amdgcn_wmma_f32_16x16x32_bf16(
            false, afrag[i], false, bfrag[j], (short)0, acc[i][j], false, false);

    // B pipeline: stash pre-loaded row into the next buffer (after compute so
    // the global-load wait sits behind the WMMAs), then start the s+2 load.
    if (s + 1 < nsteps) {
      store_b(p ^ 1, bnext);
      if (s + 2 < nsteps) bnext = load_b((s + 2) * BK);
    }

    wait_asynccnt0();   // next buffer's async DMA complete (this wave)
    __syncthreads();    // ..and visible across all waves; readers of p done
  }

  const int half = lane >> 4;
  const int cn = lane & 15;
#pragma unroll
  for (int i = 0; i < 2; ++i)
#pragma unroll
    for (int j = 0; j < 2; ++j)
#pragma unroll
      for (int e = 0; e < 8; ++e) {
        int m = m0 + wm + i * 16 + half * 8 + e;
        int n = n0 + wn + j * 16 + cn;
        epi(m, n, z, acc[i][j][e]);
      }
}

// ---------------------------------------------------------------------------
// Epilogue functors
// ---------------------------------------------------------------------------
struct EpiQKV {  // qkv GEMM: +bias, scatter q (bf16), k (bf16 transposed + present), v
  const float* bias;      // [3D]
  __bf16* qbf;            // [B,H,S,hd]
  __bf16* kT;             // [B,H,hd,S]  (so scores-GEMM B operand is row-major)
  __bf16* vbf;            // [B,H,S,hd]
  float* present;         // [B,2,H,S,hd]
  __device__ void operator()(int m, int n, int, float v) const {
    v += bias[n];
    const int b = m >> 11, s = m & (SEQ - 1);
    const int which = n >> 10, col = n & (DMODEL - 1);
    const int h = col >> 6, d = col & (HD - 1);
    const size_t bh = (size_t)(b * NHEAD + h);
    if (which == 0) {
      qbf[(bh * SEQ + s) * HD + d] = (__bf16)v;
    } else if (which == 1) {
      kT[(bh * HD + d) * SEQ + s] = (__bf16)v;
      present[((((size_t)b * 2 + 0) * NHEAD + h) * SEQ + s) * HD + d] = v;
    } else {
      vbf[(bh * SEQ + s) * HD + d] = (__bf16)v;
      present[((((size_t)b * 2 + 1) * NHEAD + h) * SEQ + s) * HD + d] = v;
    }
  }
};

struct EpiScores {  // QK^T: scale 1/sqrt(hd), causal mask, pre-softmax scores
  float* wout;      // [BH,S,S]
  __device__ void operator()(int m, int n, int z, float v) const {
    wout[((size_t)z * SEQ + m) * SEQ + n] = (n <= m) ? v * 0.125f : -1e10f;
  }
};

struct EpiAV {      // A = softmax(w) @ V -> merged [B,S,D] layout, bf16
  __bf16* abf;
  __device__ void operator()(int m, int n, int z, float v) const {
    const int b = z >> 4, h = z & (NHEAD - 1);
    abf[((size_t)(b * SEQ + m)) * DMODEL + h * HD + n] = (__bf16)v;
  }
};

struct EpiMerge {   // +bias +x residual -> xa (fp32)
  const float* bias;
  const float* x;
  float* xa;
  __device__ void operator()(int m, int n, int, float v) const {
    const size_t i = (size_t)m * DMODEL + n;
    xa[i] = v + bias[n] + x[i];
  }
};

struct EpiMLP1 {    // +bias, exact GELU -> bf16
  const float* bias;
  __bf16* m1;
  __device__ void operator()(int m, int n, int, float v) const {
    float u = v + bias[n];
    float g = 0.5f * u * (1.0f + erff(u * 0.70710678118654752f));
    m1[(size_t)m * (4 * DMODEL) + n] = (__bf16)g;
  }
};

struct EpiMLP2 {    // +bias +xa residual -> xm (fp32 output)
  const float* bias;
  const float* xa;
  float* xm;
  __device__ void operator()(int m, int n, int, float v) const {
    const size_t i = (size_t)m * DMODEL + n;
    xm[i] = v + bias[n] + xa[i];
  }
};

// ---------------------------------------------------------------------------
// Host launch
// ---------------------------------------------------------------------------
extern "C" void kernel_launch(void* const* d_in, const int* in_sizes, int n_in,
                              void* d_out, int out_size, void* d_ws, size_t ws_size,
                              hipStream_t stream) {
  (void)in_sizes; (void)n_in; (void)out_size; (void)ws_size;
  const float* x       = (const float*)d_in[0];
  const float* qkv_w   = (const float*)d_in[1];
  const float* qkv_b   = (const float*)d_in[2];
  const float* merge_w = (const float*)d_in[3];
  const float* merge_b = (const float*)d_in[4];
  const float* mlp1_w  = (const float*)d_in[5];
  const float* mlp1_b  = (const float*)d_in[6];
  const float* mlp2_w  = (const float*)d_in[7];
  const float* mlp2_b  = (const float*)d_in[8];
  const float* n1_g    = (const float*)d_in[9];
  const float* n1_b    = (const float*)d_in[10];
  const float* n2_g    = (const float*)d_in[11];
  const float* n2_b    = (const float*)d_in[12];

  // Output regions (flat, in return order): xm | present | w
  float* out_xm      = (float*)d_out;
  float* out_present = out_xm + (size_t)ROWS * DMODEL;                 // 4Mi f32
  float* out_w       = out_present + (size_t)2 * ROWS * DMODEL;        // 8Mi f32

  // Workspace bump allocator (256B aligned)
  char* wp = (char*)d_ws;
  auto alloc = [&](size_t bytes) -> char* {
    char* p = wp;
    wp += (bytes + 255) & ~(size_t)255;
    return p;
  };
  __bf16* wqkv  = (__bf16*)alloc((size_t)DMODEL * 3 * DMODEL * 2);
  __bf16* wmrg  = (__bf16*)alloc((size_t)DMODEL * DMODEL * 2);
  __bf16* wm1   = (__bf16*)alloc((size_t)DMODEL * 4 * DMODEL * 2);
  __bf16* wm2   = (__bf16*)alloc((size_t)4 * DMODEL * DMODEL * 2);
  __bf16* xn1   = (__bf16*)alloc((size_t)ROWS * DMODEL * 2);
  __bf16* qbf   = (__bf16*)alloc((size_t)BH * SEQ * HD * 2);
  __bf16* kT    = (__bf16*)alloc((size_t)BH * HD * SEQ * 2);
  __bf16* vbf   = (__bf16*)alloc((size_t)BH * SEQ * HD * 2);
  __bf16* abf   = (__bf16*)alloc((size_t)ROWS * DMODEL * 2);
  float*  xa    = (float*)alloc((size_t)ROWS * DMODEL * 4);
  __bf16* xn2   = (__bf16*)alloc((size_t)ROWS * DMODEL * 2);
  __bf16* m1    = (__bf16*)alloc((size_t)ROWS * 4 * DMODEL * 2);

  auto cast = [&](const float* src, __bf16* dst, int n) {
    cast_f32_to_bf16<<<(n / 4 + 255) / 256, 256, 0, stream>>>(src, dst, n);
  };
  // 1) one-time weight casts (fp32 -> bf16)
  cast(qkv_w,  wqkv, DMODEL * 3 * DMODEL);
  cast(merge_w, wmrg, DMODEL * DMODEL);
  cast(mlp1_w, wm1, DMODEL * 4 * DMODEL);
  cast(mlp2_w, wm2, 4 * DMODEL * DMODEL);

  // 2) LN1
  layernorm_to_bf16<<<ROWS, 256, 0, stream>>>(x, n1_g, n1_b, xn1);

  // 3) qkv GEMM [4096x1024]@[1024x3072]
  gemm_bf16_wmma<__bf16, EpiQKV><<<dim3(3 * DMODEL / BN, ROWS / BM, 1), 256, 0, stream>>>(
      xn1, DMODEL, 0, wqkv, 3 * DMODEL, 0, DMODEL,
      EpiQKV{qkv_b, qbf, kT, vbf, out_present});

  // 4) scores = Q @ K^T * 1/8, causal mask (batched over 32 heads)
  gemm_bf16_wmma<__bf16, EpiScores><<<dim3(SEQ / BN, SEQ / BM, BH), 256, 0, stream>>>(
      qbf, HD, (long long)SEQ * HD, kT, SEQ, (long long)HD * SEQ, HD,
      EpiScores{out_w});

  // 5) softmax rows (in place on output w)
  softmax_rows<<<BH * SEQ, 256, 0, stream>>>(out_w);

  // 6) A = w @ V (A is fp32, converted during LDS staging)
  gemm_bf16_wmma<float, EpiAV><<<dim3(HD / BN, SEQ / BM, BH), 256, 0, stream>>>(
      out_w, SEQ, (long long)SEQ * SEQ, vbf, HD, (long long)SEQ * HD, SEQ,
      EpiAV{abf});

  // 7) merge GEMM + residual
  gemm_bf16_wmma<__bf16, EpiMerge><<<dim3(DMODEL / BN, ROWS / BM, 1), 256, 0, stream>>>(
      abf, DMODEL, 0, wmrg, DMODEL, 0, DMODEL,
      EpiMerge{merge_b, x, xa});

  // 8) LN2
  layernorm_to_bf16<<<ROWS, 256, 0, stream>>>(xa, n2_g, n2_b, xn2);

  // 9) mlp1 + GELU
  gemm_bf16_wmma<__bf16, EpiMLP1><<<dim3(4 * DMODEL / BN, ROWS / BM, 1), 256, 0, stream>>>(
      xn2, DMODEL, 0, wm1, 4 * DMODEL, 0, DMODEL,
      EpiMLP1{mlp1_b, m1});

  // 10) mlp2 + residual -> xm
  gemm_bf16_wmma<__bf16, EpiMLP2><<<dim3(DMODEL / BN, ROWS / BM, 1), 256, 0, stream>>>(
      m1, 4 * DMODEL, 0, wm2, DMODEL, 0, 4 * DMODEL,
      EpiMLP2{mlp2_b, xa, out_xm});
}
